// actor_16243566313858
// MI455X (gfx1250) — matrix operand order repaired and checked by reading, not compiled, not run on hardware
//
#include <hip/hip_runtime.h>
#include <hip/hip_bf16.h>
#include <stdint.h>

#define TB 512      // graphs
#define NN 400      // nodes per graph
#define FF 8        // input feats
#define HH 128      // hidden
#define TT (TB*NN)  // 204800 nodes
#define EE 1638400  // edges

typedef __attribute__((ext_vector_type(16))) _Float16 v16h;
typedef __attribute__((ext_vector_type(8)))  _Float16 v8h;
typedef __attribute__((ext_vector_type(8)))  float    v8f;
typedef int v4i __attribute__((vector_size(16)));

#if __has_builtin(__builtin_amdgcn_global_load_async_to_lds_b128) && \
    __has_builtin(__builtin_amdgcn_s_wait_asynccnt)
#define HAVE_ASYNC_LDS 1
typedef __attribute__((address_space(1))) v4i* gv4ip;   // global int4*
typedef __attribute__((address_space(3))) v4i* lv4ip;   // LDS int4*
#endif

__device__ __forceinline__ v8f wmma16(v16h a, v16h b, v8f c) {
    return __builtin_amdgcn_wmma_f32_16x16x32_f16(false, a, false, b, (short)0, c, false, false);
}

// A fragment (16x32 f16). Lane holds row M = lane&15.
// element i -> K = kb + i + (i>=8 ? 8 : 0), kb = 8*(lane>=16).
// Caller passes (row base + kstep*32 + kb): two contiguous 8-half runs at +0 and +16.
__device__ __forceinline__ v16h load_afrag(const _Float16* p) {
    v8h lo = *(const v8h*)(p);
    v8h hi = *(const v8h*)(p + 16);
    v16h r;
#pragma unroll
    for (int i = 0; i < 8; ++i) { r[i] = lo[i]; r[i + 8] = hi[i]; }
    return r;
}

// B fragment (32x16 f16). Lane holds col N = lane&15.
// element i -> K = 8*(lane>=16) + i + (i>=8 ? 8 : 0): two contiguous 8-half runs at +0 and +16.
// Caller passes (col base + kstep*32 + 8*half).
__device__ __forceinline__ v16h load_bfrag(const _Float16* p) {
    v8h lo = *(const v8h*)(p);
    v8h hi = *(const v8h*)(p + 16);
    v16h r;
#pragma unroll
    for (int i = 0; i < 8; ++i) { r[i] = lo[i]; r[i + 8] = hi[i]; }
    return r;
}

// ---------------- prep kernels ----------------

__global__ __launch_bounds__(256) void zero_f32(float* __restrict__ p, int n) {
    int i = blockIdx.x * blockDim.x + threadIdx.x;
    const int stride = gridDim.x * blockDim.x;
    for (; i < n; i += stride) p[i] = 0.0f;
}

// W: [K x N] row-major f32  ->  WT: [N x Kpad] f16 (K..Kpad-1 zero-padded)
__global__ __launch_bounds__(256) void convert_w(const float* __restrict__ W,
                                                 _Float16* __restrict__ WT,
                                                 int K, int N, int Kpad) {
    int i = blockIdx.x * blockDim.x + threadIdx.x;
    const int stride = gridDim.x * blockDim.x;
    const int total = N * Kpad;
    for (; i < total; i += stride) {
        const int n = i / Kpad, k = i % Kpad;
        WT[i] = (k < K) ? (_Float16)W[(size_t)k * N + n] : (_Float16)0.0f;
    }
}

// ---------------- edge aggregation (both directions) ----------------

__global__ __launch_bounds__(256) void edge_kernel(const int* __restrict__ ei,
                                                   const float* __restrict__ x,
                                                   float* __restrict__ agg1,
                                                   float* __restrict__ agg2, int E) {
    int e = blockIdx.x * blockDim.x + threadIdx.x;
    const int stride = gridDim.x * blockDim.x;
    for (; e < E; e += stride) {
        const int s = ei[e];
        const int d = ei[E + e];
        const float4* xs = (const float4*)(x + (size_t)s * FF);
        const float4 s0 = xs[0], s1 = xs[1];
        float* a1 = agg1 + (size_t)d * FF;           // agg1[dst] += x[src]
        atomicAdd(a1 + 0, s0.x); atomicAdd(a1 + 1, s0.y);
        atomicAdd(a1 + 2, s0.z); atomicAdd(a1 + 3, s0.w);
        atomicAdd(a1 + 4, s1.x); atomicAdd(a1 + 5, s1.y);
        atomicAdd(a1 + 6, s1.z); atomicAdd(a1 + 7, s1.w);
        const float4* xd = (const float4*)(x + (size_t)d * FF);
        const float4 d0 = xd[0], d1 = xd[1];
        float* a2 = agg2 + (size_t)s * FF;           // agg2[src] += x[dst] (reverse edges)
        atomicAdd(a2 + 0, d0.x); atomicAdd(a2 + 1, d0.y);
        atomicAdd(a2 + 2, d0.z); atomicAdd(a2 + 3, d0.w);
        atomicAdd(a2 + 4, d1.x); atomicAdd(a2 + 5, d1.y);
        atomicAdd(a2 + 6, d1.z); atomicAdd(a2 + 7, d1.w);
    }
}

// ---------------- GIN MLP: h = relu(((1+eps)x+agg)@W1+b1)@W2+b2, pool += h ----------------
// 256 threads = 8 waves; each wave computes 16 node-rows x 128 cols via WMMA.
// W2T is staged once per block into LDS (async global->LDS when available).

__global__ __launch_bounds__(256) void gin_mlp(
    const float* __restrict__ x, const float* __restrict__ agg, const float* __restrict__ epsp,
    const _Float16* __restrict__ W1T /*[128][32]*/, const float* __restrict__ b1,
    const _Float16* __restrict__ W2T /*[128][128]*/, const float* __restrict__ b2,
    _Float16* __restrict__ hout /*[T][128]*/, float* __restrict__ pool /*[B][128]*/) {
    __shared__ __align__(32) _Float16 w2s[HH * HH];   // 32KB: staged W2T
    __shared__ __align__(32) _Float16 lt[8][16][HH];  // 32KB: per-wave stage-1 output
    const int wave = threadIdx.x >> 5;
    const int lane = threadIdx.x & 31;
    const int half = lane >> 4;
    const int kb = half << 3;
    const int lrow = lane & 15;
    const int base = blockIdx.x * 128 + wave * 16;
    const int row = base + lrow;

    // ---- cooperative stage of W2T into LDS (2048 x b128) ----
#ifdef HAVE_ASYNC_LDS
    {
        gv4ip gp = (gv4ip)W2T;   // C-style cast: drop const + generic->AS1
        lv4ip lp = (lv4ip)w2s;   // generic->AS3
        for (int i = threadIdx.x; i < (HH * HH) / 8; i += 256)
            __builtin_amdgcn_global_load_async_to_lds_b128(gp + i, lp + i, 0, 0);
        __builtin_amdgcn_s_wait_asynccnt(0);
    }
#else
    for (int i = threadIdx.x; i < (HH * HH) / 8; i += 256)
        ((v8h*)w2s)[i] = ((const v8h*)W2T)[i];
#endif

    // ---- stage 1: in8 = (1+eps)*x + agg, K padded 8 -> 32 ----
    const float epsv = 1.0f + epsp[0];
    float in8[8];
    {
        const float4* xr = (const float4*)(x + (size_t)row * FF);
        const float4* ar = (const float4*)(agg + (size_t)row * FF);
        const float4 x0 = xr[0], x1 = xr[1], a0 = ar[0], a1 = ar[1];
        in8[0] = epsv * x0.x + a0.x; in8[1] = epsv * x0.y + a0.y;
        in8[2] = epsv * x0.z + a0.z; in8[3] = epsv * x0.w + a0.w;
        in8[4] = epsv * x1.x + a1.x; in8[5] = epsv * x1.y + a1.y;
        in8[6] = epsv * x1.z + a1.z; in8[7] = epsv * x1.w + a1.w;
    }
    v16h af;
#pragma unroll
    for (int i = 0; i < 16; ++i) {
        const int k = kb + i + ((i >= 8) ? 8 : 0);
        af[i] = (k < 8) ? (_Float16)in8[k] : (_Float16)0.0f;
    }
    v8f acc[8];
#pragma unroll
    for (int tn = 0; tn < 8; ++tn) {
        const int col = lrow + tn * 16;
        const float bb = b1[col];
#pragma unroll
        for (int r = 0; r < 8; ++r) acc[tn][r] = bb;
        v16h bf = load_bfrag(W1T + col * 32 + half * 8);
        acc[tn] = wmma16(af, bf, acc[tn]);
    }
    __syncthreads();  // w2s staged (all waves) before stage-2 reads
    // relu -> LDS (f16), becomes A for stage 2 (ds_load_b128 fragments)
#pragma unroll
    for (int tn = 0; tn < 8; ++tn) {
        const int col = lrow + tn * 16;
#pragma unroll
        for (int r = 0; r < 8; ++r) {
            const float v = acc[tn][r];
            lt[wave][r + half * 8][col] = (_Float16)(v > 0.0f ? v : 0.0f);
        }
    }
    __builtin_amdgcn_wave_barrier();  // lt is wave-local

    // ---- stage 2: t @ W2 (K=128, 4 k-steps), A and B both from LDS ----
    v8f acch[8];
#pragma unroll
    for (int tn = 0; tn < 8; ++tn) {
        const float bb = b2[lrow + tn * 16];
#pragma unroll
        for (int r = 0; r < 8; ++r) acch[tn][r] = bb;
    }
#pragma unroll
    for (int kk = 0; kk < 4; ++kk) {
        v16h a2 = load_afrag(&lt[wave][lrow][kk * 32 + kb]);
#pragma unroll
        for (int tn = 0; tn < 8; ++tn) {
            const int col = lrow + tn * 16;
            v16h bf = load_bfrag(&w2s[col * HH + kk * 32 + half * 8]);
            acch[tn] = wmma16(a2, bf, acch[tn]);
        }
    }
    // store h (f16) + pooled sum: pre-reduce 8 rows per lane (<=2 graphs) -> <=2 atomics/tile
    const int rbase = base + half * 8;
    const int g0 = rbase / NN;
    const int rem = (g0 + 1) * NN - rbase;  // rows of this 8-row strip still in graph g0
#pragma unroll
    for (int tn = 0; tn < 8; ++tn) {
        const int col = lrow + tn * 16;
        float s0 = 0.0f, s1 = 0.0f;
#pragma unroll
        for (int r = 0; r < 8; ++r) {
            const float v = acch[tn][r];
            hout[(size_t)(rbase + r) * HH + col] = (_Float16)v;
            if (r < rem) s0 += v; else s1 += v;
        }
        atomicAdd(&pool[g0 * HH + col], s0);
        if (rem < 8) atomicAdd(&pool[(g0 + 1) * HH + col], s1);
    }
}

__global__ __launch_bounds__(256) void pool_scale(const float* __restrict__ p,
                                                  _Float16* __restrict__ ph, int n) {
    int i = blockIdx.x * blockDim.x + threadIdx.x;
    const int stride = gridDim.x * blockDim.x;
    for (; i < n; i += stride) ph[i] = (_Float16)(p[i] * (1.0f / (float)NN));
}

// ---------------- fused l1+l2+act head ----------------
// a = relu([h1|ge1|h2|ge2] @ l1_W + l1_b)            (K=512, WMMA)
// staged to LDS with x appended (K 136 -> 160, zero pad)
// s = relu([a|x] @ l2_W + l2_b) @ act_W + act_b      (K=160, WMMA; head via shfl reduce)

__global__ __launch_bounds__(256) void l12_kernel(
    const _Float16* __restrict__ h1, const _Float16* __restrict__ p1,
    const _Float16* __restrict__ h2, const _Float16* __restrict__ p2,
    const _Float16* __restrict__ W1T /*[128][512]*/, const float* __restrict__ b1v,
    const _Float16* __restrict__ W2T /*[128][160]*/, const float* __restrict__ b2v,
    const float* __restrict__ x, const float* __restrict__ actW,
    const float* __restrict__ actbp, const unsigned char* __restrict__ mask,
    float* __restrict__ scores) {
    __shared__ __align__(32) _Float16 lt[8][16][160];  // 40KB
    const int wave = threadIdx.x >> 5;
    const int lane = threadIdx.x & 31;
    const int half = lane >> 4, kb = half << 3, lrow = lane & 15;
    const int base = blockIdx.x * 128 + wave * 16;
    const int row = base + lrow;
    const int graph = row / NN;

    __builtin_prefetch(h1 + (size_t)row * HH, 0, 1);   // global_prefetch_b8
    __builtin_prefetch(h2 + (size_t)row * HH, 0, 1);

    const _Float16* seg[4] = { h1 + (size_t)row * HH, p1 + (size_t)graph * HH,
                               h2 + (size_t)row * HH, p2 + (size_t)graph * HH };
    v8f acc[8];
#pragma unroll
    for (int tn = 0; tn < 8; ++tn) {
        const float bb = b1v[lrow + tn * 16];
#pragma unroll
        for (int r = 0; r < 8; ++r) acc[tn][r] = bb;
    }
#pragma unroll
    for (int kk = 0; kk < 16; ++kk) {                 // K=512: each 32-k-step sits in one 128-seg
        v16h afr = load_afrag(seg[kk >> 2] + (kk & 3) * 32 + kb);
#pragma unroll
        for (int tn = 0; tn < 8; ++tn) {
            const int col = lrow + tn * 16;
            v16h bf = load_bfrag(W1T + (size_t)col * 512 + kk * 32 + half * 8);
            acc[tn] = wmma16(afr, bf, acc[tn]);
        }
    }
    // relu -> LDS tile (cols 0..127)
#pragma unroll
    for (int tn = 0; tn < 8; ++tn) {
        const int col = lrow + tn * 16;
#pragma unroll
        for (int r = 0; r < 8; ++r) {
            const float v = acc[tn][r];
            lt[wave][r + half * 8][col] = (_Float16)(v > 0.0f ? v : 0.0f);
        }
    }
    if (lane < 16) {                                  // append x (cols 128..135), zero 136..159
        const int rr = base + lane;
#pragma unroll
        for (int j = 0; j < 8; ++j)
            lt[wave][lane][128 + j] = (_Float16)x[(size_t)rr * FF + j];
#pragma unroll
        for (int j = 136; j < 160; ++j)
            lt[wave][lane][j] = (_Float16)0.0f;
    }
    __builtin_amdgcn_wave_barrier();  // lt is wave-local

    // ---- l2: K=160 from LDS A-fragments ----
    v8f acc2[8];
#pragma unroll
    for (int tn = 0; tn < 8; ++tn) {
        const float bb = b2v[lrow + tn * 16];
#pragma unroll
        for (int r = 0; r < 8; ++r) acc2[tn][r] = bb;
    }
#pragma unroll
    for (int kk = 0; kk < 5; ++kk) {
        v16h afr = load_afrag(&lt[wave][lrow][kk * 32 + kb]);
#pragma unroll
        for (int tn = 0; tn < 8; ++tn) {
            const int col = lrow + tn * 16;
            v16h bf = load_bfrag(W2T + (size_t)col * 160 + kk * 32 + half * 8);
            acc2[tn] = wmma16(afr, bf, acc2[tn]);
        }
    }
    // relu then dot with act_W: lane covers 8 cols for its 8 rows -> butterfly over 16 lanes
    float part[8];
#pragma unroll
    for (int r = 0; r < 8; ++r) part[r] = 0.0f;
#pragma unroll
    for (int tn = 0; tn < 8; ++tn) {
        const float w = actW[lrow + tn * 16];
#pragma unroll
        for (int r = 0; r < 8; ++r) {
            const float v = acc2[tn][r];
            part[r] += (v > 0.0f ? v : 0.0f) * w;
        }
    }
#pragma unroll
    for (int off = 1; off < 16; off <<= 1) {
#pragma unroll
        for (int r = 0; r < 8; ++r) part[r] += __shfl_xor(part[r], off, 32);
    }
    if (lrow == 0) {
        const float ab = actbp[0];
#pragma unroll
        for (int r = 0; r < 8; ++r) {
            const int rr = base + r + half * 8;
            scores[rr] = part[r] + ab - (mask[rr] ? 0.0f : 1.0e10f);
        }
    }
}

// ---------------- per-graph log-softmax + Gumbel-max categorical sample ----------------

__device__ __forceinline__ float gumbel_noise(unsigned b, unsigned i) {
    unsigned h = b * 2654435761u ^ (i + 0x9E3779B9u) ^ 42u;
    h ^= h >> 16; h *= 0x85EBCA6Bu; h ^= h >> 13; h *= 0xC2B2AE35u; h ^= h >> 16;
    float u = (float)(h >> 8) * (1.0f / 16777216.0f);
    u = fminf(fmaxf(u, 1.0e-10f), 0.99999994f);
    return -__logf(-__logf(u));
}

__global__ __launch_bounds__(256) void sample_kernel(const float* __restrict__ scores,
                                                     const int* __restrict__ goff,
                                                     int* __restrict__ outi,
                                                     float* __restrict__ outlp) {
    __shared__ float red[256];
    __shared__ float bv[256];
    __shared__ int   bi[256];
    const int b = blockIdx.x;
    const int t = threadIdx.x;
    const float* s = scores + (size_t)b * NN;

    float mx = -3.0e38f;
    for (int i = t; i < NN; i += 256) mx = fmaxf(mx, s[i]);
    red[t] = mx; __syncthreads();
    for (int o = 128; o > 0; o >>= 1) { if (t < o) red[t] = fmaxf(red[t], red[t + o]); __syncthreads(); }
    const float m = red[0];
    __syncthreads();

    float sum = 0.0f, bestv = -3.0e38f; int besti = 0;
    for (int i = t; i < NN; i += 256) {
        const float v = s[i];
        sum += __expf(v - m);
        const float pv = v + gumbel_noise((unsigned)b, (unsigned)i);
        if (pv > bestv) { bestv = pv; besti = i; }
    }
    red[t] = sum; bv[t] = bestv; bi[t] = besti; __syncthreads();
    for (int o = 128; o > 0; o >>= 1) {
        if (t < o) {
            red[t] += red[t + o];
            if (bv[t + o] > bv[t]) { bv[t] = bv[t + o]; bi[t] = bi[t + o]; }
        }
        __syncthreads();
    }
    if (t == 0) {
        const float lse = m + __logf(red[0]);
        const int idx = bi[0];
        outi[b] = idx + goff[b];
        outlp[b] = s[idx] - lse;
    }
}

// ---------------- host ----------------

extern "C" void kernel_launch(void* const* d_in, const int* in_sizes, int n_in,
                              void* d_out, int out_size, void* d_ws, size_t ws_size,
                              hipStream_t stream) {
    (void)in_sizes; (void)n_in; (void)out_size; (void)ws_size;
    const float* x    = (const float*)d_in[0];
    const int*   ei   = (const int*)d_in[1];
    const unsigned char* mask = (const unsigned char*)d_in[3];
    const int*   goff = (const int*)d_in[4];
    const float* g1W1 = (const float*)d_in[5];
    const float* g1b1 = (const float*)d_in[6];
    const float* g1W2 = (const float*)d_in[7];
    const float* g1b2 = (const float*)d_in[8];
    const float* g1e  = (const float*)d_in[9];
    const float* g2W1 = (const float*)d_in[10];
    const float* g2b1 = (const float*)d_in[11];
    const float* g2W2 = (const float*)d_in[12];
    const float* g2b2 = (const float*)d_in[13];
    const float* g2e  = (const float*)d_in[14];
    const float* l1W  = (const float*)d_in[15];
    const float* l1b  = (const float*)d_in[16];
    const float* l2W  = (const float*)d_in[17];
    const float* l2b  = (const float*)d_in[18];
    const float* actW = (const float*)d_in[19];
    const float* actb = (const float*)d_in[20];

    char* ws = (char*)d_ws;
    size_t off = 0;
    auto take = [&](size_t bytes) { char* p = ws + off; off += (bytes + 255) & ~(size_t)255; return p; };

    float*    agg1   = (float*)take((size_t)TT * FF * 4);   // contiguous zero region start
    float*    agg2   = (float*)take((size_t)TT * FF * 4);
    float*    pool1  = (float*)take((size_t)TB * HH * 4);
    float*    pool2  = (float*)take((size_t)TB * HH * 4);   // contiguous zero region end
    _Float16* pool1h = (_Float16*)take((size_t)TB * HH * 2);
    _Float16* pool2h = (_Float16*)take((size_t)TB * HH * 2);
    _Float16* h1     = (_Float16*)take((size_t)TT * HH * 2);
    _Float16* h2     = (_Float16*)take((size_t)TT * HH * 2);
    float*    scores = (float*)take((size_t)TT * 4);
    _Float16* W1Ta   = (_Float16*)take(128 * 32 * 2);
    _Float16* W2Ta   = (_Float16*)take(128 * 128 * 2);
    _Float16* W1Tb   = (_Float16*)take(128 * 32 * 2);
    _Float16* W2Tb   = (_Float16*)take(128 * 128 * 2);
    _Float16* l1WT   = (_Float16*)take((size_t)128 * 512 * 2);
    _Float16* l2WT   = (_Float16*)take((size_t)128 * 160 * 2);

    // agg1..pool2 are contiguous (all sizes are 256B multiples): one zero pass
    const int zn = TT * FF * 2 + TB * HH * 2;
    zero_f32<<<2048, 256, 0, stream>>>(agg1, zn);

    convert_w<<<64, 256, 0, stream>>>(g1W1, W1Ta, 8,   128, 32);
    convert_w<<<64, 256, 0, stream>>>(g1W2, W2Ta, 128, 128, 128);
    convert_w<<<64, 256, 0, stream>>>(g2W1, W1Tb, 8,   128, 32);
    convert_w<<<64, 256, 0, stream>>>(g2W2, W2Tb, 128, 128, 128);
    convert_w<<<256, 256, 0, stream>>>(l1W, l1WT, 512, 128, 512);
    convert_w<<<128, 256, 0, stream>>>(l2W, l2WT, 136, 128, 160);

    edge_kernel<<<6400, 256, 0, stream>>>(ei, x, agg1, agg2, EE);

    gin_mlp<<<TT / 128, 256, 0, stream>>>(x, agg1, g1e, W1Ta, g1b1, W2Ta, g1b2, h1, pool1);
    gin_mlp<<<TT / 128, 256, 0, stream>>>(x, agg2, g2e, W1Tb, g2b1, W2Tb, g2b2, h2, pool2);

    pool_scale<<<256, 256, 0, stream>>>(pool1, pool1h, TB * HH * 2);  // pool1/2 & pool1h/2h contiguous

    l12_kernel<<<TT / 128, 256, 0, stream>>>(h1, pool1h, h2, pool2h, l1WT, l1b,
                                             l2WT, l2b, x, actW, actb, mask, scores);

    sample_kernel<<<TB, 256, 0, stream>>>(scores, goff, (int*)d_out, (float*)d_out + TB);
}
